// NLayerDeeperGCN_2035814498365
// MI455X (gfx1250) — matrix operand order repaired and checked by reading, not compiled, not run on hardware
//
#include <hip/hip_runtime.h>
#include <hip/hip_bf16.h>
#include <math.h>

#define N_NODES 40000
#define N_EDGES 640000
#define F_DIM   128
#define C_DIM   20
#define HID_DIM 256
#define N_LAYERS 3
#define GEN_EPS 1e-7f
#define LN_EPS  1e-5f

typedef __attribute__((ext_vector_type(16))) __bf16 v16bf;
typedef __attribute__((ext_vector_type(8)))  float  v8f;

// float -> bf16 bits, round-to-nearest-even
__device__ __forceinline__ unsigned short f2bf_u(float f) {
    unsigned u = __float_as_uint(f);
    unsigned r = (u + 0x7FFFu + ((u >> 16) & 1u)) >> 16;
    return (unsigned short)r;
}

// order-preserving encodings for float atomicMax on uint
__device__ __forceinline__ unsigned f2o(float f) {
    unsigned u = __float_as_uint(f);
    return (u & 0x80000000u) ? ~u : (u | 0x80000000u);
}
__device__ __forceinline__ float o2f(unsigned u) {
    return (u & 0x80000000u) ? __uint_as_float(u & 0x7FFFFFFFu)
                             : __uint_as_float(~u);
}

__device__ __forceinline__ float wave_reduce_sum(float v) {
    #pragma unroll
    for (int m = 16; m >= 1; m >>= 1) v += __shfl_xor(v, m, 32);
    return v;
}

// A-matrix (16x32 bf16) fragment K index for (lane-group g, vector elem j)
__device__ __forceinline__ int a_frag_k(int g, int j) {
    int v = j >> 1, hh = j & 1;
    return (v < 4) ? (g * 8 + 2 * v + hh) : (16 + g * 8 + 2 * (v - 4) + hh);
}

// ---------------- embedding gather: h[n,f] = emb[ids[n], f] ----------------
__global__ void gather_kernel(const int* __restrict__ ids,
                              const float* __restrict__ emb,
                              float* __restrict__ h) {
    int idx = blockIdx.x * blockDim.x + threadIdx.x;   // N*32 threads, 4 floats each
    int n = idx >> 5;
    int fc = (idx & 31) << 2;
    if (n >= N_NODES) return;
    const float4 v = *(const float4*)(emb + (long)ids[n] * F_DIM + fc);
    *(float4*)(h + (long)n * F_DIM + fc) = v;
}

// ---- pack weight [kTotal x nCols] (row-major, fp32) into bf16 WMMA --------
// ---- B-fragment order: wpk[((ct*numK + kk)*32 + lane)*16 + j] -------------
__global__ void pack_w_kernel(const float* __restrict__ w,
                              unsigned short* __restrict__ wpk,
                              int kTotal, int nCols) {
    int idx = blockIdx.x * blockDim.x + threadIdx.x;
    if (idx >= kTotal * nCols) return;
    int j    = idx & 15;
    int lane = (idx >> 4) & 31;
    int frag = idx >> 9;                 // ct*numK + kk
    int numK = kTotal >> 5;
    int ct = frag / numK, kk = frag - ct * numK;
    int g = lane >> 4, nc = lane & 15;
    int K   = kk * 32 + a_frag_k(g, j);  // B layout: same K order as A fragment
    int col = ct * 16 + nc;
    wpk[idx] = f2bf_u(w[(long)K * nCols + col]);
}

// -------- per-node LayerNorm(F=128) + ReLU: one wave per node --------------
__global__ void ln_relu_kernel(const float* __restrict__ h,
                               const float* __restrict__ g,
                               const float* __restrict__ b,
                               float* __restrict__ xn) {
    int wid  = (blockIdx.x * blockDim.x + threadIdx.x) >> 5;
    int lane = threadIdx.x & 31;
    if (wid >= N_NODES) return;
    float4 v = ((const float4*)(h + (long)wid * F_DIM))[lane];
    float s  = v.x + v.y + v.z + v.w;
    float sq = v.x*v.x + v.y*v.y + v.z*v.z + v.w*v.w;
    s  = wave_reduce_sum(s);
    sq = wave_reduce_sum(sq);
    float mean = s * (1.0f / F_DIM);
    float var  = sq * (1.0f / F_DIM) - mean * mean;
    float rstd = rsqrtf(var + LN_EPS);
    int c = lane * 4;
    float4 o;
    o.x = fmaxf((v.x - mean) * rstd * g[c + 0] + b[c + 0], 0.0f);
    o.y = fmaxf((v.y - mean) * rstd * g[c + 1] + b[c + 1], 0.0f);
    o.z = fmaxf((v.z - mean) * rstd * g[c + 2] + b[c + 2], 0.0f);
    o.w = fmaxf((v.w - mean) * rstd * g[c + 3] + b[c + 3], 0.0f);
    ((float4*)(xn + (long)wid * F_DIM))[lane] = o;
}

// -------- init segment buffers: m = -inf (encoded), s = ms = 0 -------------
__global__ void seg_init_kernel(unsigned* __restrict__ m,
                                float* __restrict__ s,
                                float* __restrict__ ms) {
    int i = blockIdx.x * blockDim.x + threadIdx.x;
    if (i >= N_NODES * F_DIM) return;
    m[i]  = f2o(-1e30f);
    s[i]  = 0.0f;
    ms[i] = 0.0f;
}

// -------- edge pass 1: segment max of logits = t*(relu(xn[src])+eps) -------
__global__ void edge_max_kernel(const float* __restrict__ xn,
                                const int* __restrict__ src,
                                const int* __restrict__ dst,
                                const float* __restrict__ tv,
                                unsigned* __restrict__ m) {
    int idx = blockIdx.x * blockDim.x + threadIdx.x;   // E*32 threads, 4 feats each
    int e = idx >> 5;
    if (e >= N_EDGES) return;
    int fc = (idx & 31) << 2;
    float tval = tv[0];
    int sN = src[e], dN = dst[e];
    float4 x = *(const float4*)(xn + (long)sN * F_DIM + fc);
    unsigned* mp = m + (long)dN * F_DIM + fc;
    atomicMax(mp + 0, f2o((x.x + GEN_EPS) * tval));
    atomicMax(mp + 1, f2o((x.y + GEN_EPS) * tval));
    atomicMax(mp + 2, f2o((x.z + GEN_EPS) * tval));
    atomicMax(mp + 3, f2o((x.w + GEN_EPS) * tval));
}

// -------- edge pass 2: accumulate e = exp(logit-m) and msg*e ---------------
__global__ void edge_sum_kernel(const float* __restrict__ xn,
                                const int* __restrict__ src,
                                const int* __restrict__ dst,
                                const float* __restrict__ tv,
                                const unsigned* __restrict__ m,
                                float* __restrict__ s,
                                float* __restrict__ ms) {
    int idx = blockIdx.x * blockDim.x + threadIdx.x;
    int e = idx >> 5;
    if (e >= N_EDGES) return;
    int fc = (idx & 31) << 2;
    float tval = tv[0];
    int sN = src[e], dN = dst[e];
    float4 x = *(const float4*)(xn + (long)sN * F_DIM + fc);
    long dB = (long)dN * F_DIM + fc;
    float msg0 = x.x + GEN_EPS, msg1 = x.y + GEN_EPS;
    float msg2 = x.z + GEN_EPS, msg3 = x.w + GEN_EPS;
    float e0 = __expf(msg0 * tval - o2f(m[dB + 0]));
    float e1 = __expf(msg1 * tval - o2f(m[dB + 1]));
    float e2 = __expf(msg2 * tval - o2f(m[dB + 2]));
    float e3 = __expf(msg3 * tval - o2f(m[dB + 3]));
    atomicAdd(s  + dB + 0, e0);          atomicAdd(s  + dB + 1, e1);
    atomicAdd(s  + dB + 2, e2);          atomicAdd(s  + dB + 3, e3);
    atomicAdd(ms + dB + 0, msg0 * e0);   atomicAdd(ms + dB + 1, msg1 * e1);
    atomicAdd(ms + dB + 2, msg2 * e2);   atomicAdd(ms + dB + 3, msg3 * e3);
}

// -------- hmid = xn + aggr  (in-place into xn) -----------------------------
__global__ void combine_kernel(float* __restrict__ xn,
                               const float* __restrict__ s,
                               const float* __restrict__ ms) {
    int i = blockIdx.x * blockDim.x + threadIdx.x;
    if (i >= N_NODES * F_DIM) return;
    xn[i] = xn[i] + ms[i] / (s[i] + 1e-16f);
}

// -------- fused MLP: GEMM1(bf16 WMMA) -> +b1 -> LN(256)+ReLU ->
//          GEMM2(bf16 WMMA) -> +b2 -> residual into h -----------------------
// Weights come pre-packed in fragment order; A operands staged in LDS in
// fragment order so each lane fragment is two b128 loads.
__global__ void __launch_bounds__(256)
mlp_fused_kernel(const float* __restrict__ hmid,
                 float* __restrict__ h,
                 const unsigned short* __restrict__ w1pk,
                 const float* __restrict__ b1,
                 const float* __restrict__ mg, const float* __restrict__ mb,
                 const unsigned short* __restrict__ w2pk,
                 const float* __restrict__ b2) {
    __shared__ __align__(32) unsigned short lds_a[16 * F_DIM];     // A frags (bf16)
    __shared__ float                        lds_mid[16 * HID_DIM]; // GEMM1 out fp32
    __shared__ __align__(32) unsigned short lds_mbf[16 * HID_DIM]; // GEMM2 A frags

    const int tid  = threadIdx.x;
    const int wave = tid >> 5;
    const int lane = tid & 31;
    const int g    = lane >> 4;      // lane group
    const int nc   = lane & 15;      // tile row (A) / tile col (B,C)
    const long node0 = (long)blockIdx.x * 16;

    // stage 16xF A panel as bf16 in fragment order:
    // lds_a[(kk*32 + lane')*16 + j] = A[m = lane'&15][kk*32 + a_frag_k(g',j)]
    for (int i = tid; i < 16 * F_DIM; i += 256) {
        int j  = i & 15;
        int ln = (i >> 4) & 31;
        int kk = i >> 9;
        int gg = ln >> 4, m = ln & 15;
        int K  = kk * 32 + a_frag_k(gg, j);
        lds_a[i] = f2bf_u(hmid[node0 * F_DIM + (long)m * F_DIM + K]);
    }
    __syncthreads();

    const v16bf* w1f = (const v16bf*)w1pk;   // [16 tiles][4 kk][32 lanes]
    const v16bf* w2f = (const v16bf*)w2pk;   // [8 tiles][8 kk][32 lanes]
    const v16bf* af  = (const v16bf*)lds_a;  // [4 kk][32 lanes]

    // ---- GEMM1: each wave owns 2 of 16 col tiles of HID ----
    v8f acc0 = {}; v8f acc1 = {};
    #pragma unroll
    for (int kk = 0; kk < F_DIM / 32; ++kk) {
        v16bf av  = af[kk * 32 + lane];
        v16bf bv0 = w1f[((wave * 2 + 0) * (F_DIM / 32) + kk) * 32 + lane];
        v16bf bv1 = w1f[((wave * 2 + 1) * (F_DIM / 32) + kk) * 32 + lane];
        acc0 = __builtin_amdgcn_wmma_f32_16x16x32_bf16(
            false, av, false, bv0, (short)0, acc0, false, false);
        acc1 = __builtin_amdgcn_wmma_f32_16x16x32_bf16(
            false, av, false, bv1, (short)0, acc1, false, false);
    }
    #pragma unroll
    for (int r = 0; r < 8; ++r) {
        int row = g ? (r + 8) : r;
        int c0 = (wave * 2) * 16 + nc;
        int c1 = (wave * 2 + 1) * 16 + nc;
        lds_mid[row * HID_DIM + c0] = acc0[r] + b1[c0];
        lds_mid[row * HID_DIM + c1] = acc1[r] + b1[c1];
    }
    __syncthreads();

    // ---- LayerNorm(256)+ReLU, one wave per 2 rows; write bf16 frags ----
    #pragma unroll
    for (int rr = 0; rr < 2; ++rr) {
        int row = wave * 2 + rr;
        float s = 0.0f, sq = 0.0f;
        #pragma unroll
        for (int j = 0; j < 8; ++j) {
            float x = lds_mid[row * HID_DIM + lane * 8 + j];
            s += x; sq += x * x;
        }
        s  = wave_reduce_sum(s);
        sq = wave_reduce_sum(sq);
        float mean = s * (1.0f / HID_DIM);
        float var  = sq * (1.0f / HID_DIM) - mean * mean;
        float rstd = rsqrtf(var + LN_EPS);
        #pragma unroll
        for (int j = 0; j < 8; ++j) {
            int c = lane * 8 + j;
            float x = lds_mid[row * HID_DIM + c];
            float y = fmaxf((x - mean) * rstd * mg[c] + mb[c], 0.0f);
            // inverse of A-fragment layout: (row, c) -> fragment position
            int kk2 = c >> 5, kc = c & 31;
            int gg  = (kc & 15) >> 3;
            int jj  = (kc & 7) | ((kc & 16) >> 1);
            int ln2 = gg * 16 + row;
            lds_mbf[(kk2 * 32 + ln2) * 16 + jj] = f2bf_u(y);
        }
    }
    __syncthreads();

    const v16bf* mf = (const v16bf*)lds_mbf;  // [8 kk][32 lanes]

    // ---- GEMM2: each wave owns 1 of 8 col tiles of F ----
    v8f acc = {};
    const int cb = wave * 16;
    #pragma unroll
    for (int kk = 0; kk < HID_DIM / 32; ++kk) {
        v16bf av = mf[kk * 32 + lane];
        v16bf bv = w2f[(wave * (HID_DIM / 32) + kk) * 32 + lane];
        acc = __builtin_amdgcn_wmma_f32_16x16x32_bf16(
            false, av, false, bv, (short)0, acc, false, false);
    }
    // residual: h = h + (GEMM2 + b2)
    #pragma unroll
    for (int r = 0; r < 8; ++r) {
        int row = g ? (r + 8) : r;
        int col = cb + nc;
        long idx = (node0 + row) * F_DIM + col;
        h[idx] = h[idx] + acc[r] + b2[col];
    }
}

// ----------------------------- GCNConv head --------------------------------
__global__ void xw_kernel(const float* __restrict__ h,
                          const float* __restrict__ w,
                          float* __restrict__ xw) {
    int idx = blockIdx.x * blockDim.x + threadIdx.x;
    if (idx >= N_NODES * C_DIM) return;
    int n = idx / C_DIM, c = idx % C_DIM;
    const float* hr = h + (long)n * F_DIM;
    float acc = 0.0f;
    #pragma unroll 4
    for (int k = 0; k < F_DIM; ++k) acc += hr[k] * w[k * C_DIM + c];
    xw[idx] = acc;
}

__global__ void deg_init_kernel(float* __restrict__ deg) {
    int i = blockIdx.x * blockDim.x + threadIdx.x;
    if (i < N_NODES) deg[i] = 1.0f;   // self-loop weight
}

__global__ void deg_edge_kernel(const int* __restrict__ dst,
                                const float* __restrict__ ew,
                                float* __restrict__ deg) {
    int e = blockIdx.x * blockDim.x + threadIdx.x;
    if (e < N_EDGES) atomicAdd(&deg[dst[e]], ew[e]);
}

__global__ void dis_kernel(const float* __restrict__ deg,
                           float* __restrict__ dis) {
    int i = blockIdx.x * blockDim.x + threadIdx.x;
    if (i >= N_NODES) return;
    float d = deg[i];
    dis[i] = (d > 0.0f) ? rsqrtf(d) : 0.0f;
}

__global__ void out_init_kernel(const float* __restrict__ xw,
                                const float* __restrict__ dis,
                                const float* __restrict__ gb,
                                float* __restrict__ out) {
    int idx = blockIdx.x * blockDim.x + threadIdx.x;
    if (idx >= N_NODES * C_DIM) return;
    int n = idx / C_DIM, c = idx % C_DIM;
    float d = dis[n];
    out[idx] = gb[c] + d * d * xw[idx];   // self-loop contribution + bias
}

__global__ void gcn_edge_kernel(const int* __restrict__ src,
                                const int* __restrict__ dst,
                                const float* __restrict__ ew,
                                const float* __restrict__ dis,
                                const float* __restrict__ xw,
                                float* __restrict__ out) {
    int e = blockIdx.x * blockDim.x + threadIdx.x;
    if (e >= N_EDGES) return;
    int sN = src[e], dN = dst[e];
    float coef = dis[sN] * ew[e] * dis[dN];
    const float* xr = xw + (long)sN * C_DIM;
    float* orow = out + (long)dN * C_DIM;
    #pragma unroll
    for (int c = 0; c < C_DIM; ++c) atomicAdd(orow + c, coef * xr[c]);
}

__global__ void lsm_kernel(float* __restrict__ out) {
    int n = blockIdx.x * blockDim.x + threadIdx.x;
    if (n >= N_NODES) return;
    float* r = out + (long)n * C_DIM;
    float mx = r[0];
    #pragma unroll
    for (int c = 1; c < C_DIM; ++c) mx = fmaxf(mx, r[c]);
    float sum = 0.0f;
    #pragma unroll
    for (int c = 0; c < C_DIM; ++c) sum += __expf(r[c] - mx);
    float lse = mx + __logf(sum);
    #pragma unroll
    for (int c = 0; c < C_DIM; ++c) r[c] = r[c] - lse;
}

extern "C" void kernel_launch(void* const* d_in, const int* in_sizes, int n_in,
                              void* d_out, int out_size, void* d_ws, size_t ws_size,
                              hipStream_t stream) {
    const int*   x_ids = (const int*)d_in[0];
    const int*   eidx  = (const int*)d_in[1];   // [2, E]
    const float* eattr = (const float*)d_in[2];
    const float* emb   = (const float*)d_in[3];
    const float* ln_g  = (const float*)d_in[4];
    const float* ln_b  = (const float*)d_in[5];
    const float* t_arr = (const float*)d_in[6];
    const float* w1    = (const float*)d_in[7];
    const float* b1    = (const float*)d_in[8];
    const float* mlp_g = (const float*)d_in[9];
    const float* mlp_b = (const float*)d_in[10];
    const float* w2    = (const float*)d_in[11];
    const float* b2    = (const float*)d_in[12];
    const float* gcn_w = (const float*)d_in[13];
    const float* gcn_b = (const float*)d_in[14];
    float* out = (float*)d_out;

    const int* src = eidx;
    const int* dst = eidx + N_EDGES;

    const long NF = (long)N_NODES * F_DIM;
    const int  WSZ = F_DIM * HID_DIM;           // 32768 elements per weight
    float* ws   = (float*)d_ws;
    float* h    = ws;                 // [N,F]
    float* xn   = ws + 1 * NF;        // [N,F]  (reused as hmid)
    unsigned* m = (unsigned*)(ws + 2 * NF);  // [N,F] encoded max
    float* sB   = ws + 3 * NF;        // [N,F] sum of exp
    float* msB  = ws + 4 * NF;        // [N,F] sum of msg*exp
    float* xw   = ws + 5 * NF;        // [N,C]
    float* deg  = xw + (long)N_NODES * C_DIM;  // [N]
    float* dis  = deg + N_NODES;               // [N]
    unsigned short* w1pk = (unsigned short*)(dis + N_NODES);  // 3*WSZ bf16
    unsigned short* w2pk = w1pk + (long)N_LAYERS * WSZ;       // 3*WSZ bf16

    const int TPB = 256;
    const int gNodeVec = (N_NODES * 32 + TPB - 1) / TPB;
    const int gEdgeVec = (N_EDGES * 32 + TPB - 1) / TPB;
    const int gNF      = (int)((NF + TPB - 1) / TPB);
    const int gNC      = (N_NODES * C_DIM + TPB - 1) / TPB;
    const int gN       = (N_NODES + TPB - 1) / TPB;
    const int gE       = (N_EDGES + TPB - 1) / TPB;
    const int gW       = (WSZ + TPB - 1) / TPB;

    // pre-pack all layer weights to bf16 fragment order (tiny, L2-resident)
    for (int i = 0; i < N_LAYERS; ++i) {
        pack_w_kernel<<<gW, TPB, 0, stream>>>(w1 + (long)i * WSZ,
                                              w1pk + (long)i * WSZ,
                                              F_DIM, HID_DIM);
        pack_w_kernel<<<gW, TPB, 0, stream>>>(w2 + (long)i * WSZ,
                                              w2pk + (long)i * WSZ,
                                              HID_DIM, F_DIM);
    }

    gather_kernel<<<gNodeVec, TPB, 0, stream>>>(x_ids, emb, h);

    for (int i = 0; i < N_LAYERS; ++i) {
        ln_relu_kernel<<<gNodeVec, TPB, 0, stream>>>(h, ln_g + i * F_DIM,
                                                     ln_b + i * F_DIM, xn);
        seg_init_kernel<<<gNF, TPB, 0, stream>>>(m, sB, msB);
        edge_max_kernel<<<gEdgeVec, TPB, 0, stream>>>(xn, src, dst, t_arr + i, m);
        edge_sum_kernel<<<gEdgeVec, TPB, 0, stream>>>(xn, src, dst, t_arr + i,
                                                      m, sB, msB);
        combine_kernel<<<gNF, TPB, 0, stream>>>(xn, sB, msB);
        mlp_fused_kernel<<<N_NODES / 16, TPB, 0, stream>>>(
            xn, h,
            w1pk + (long)i * WSZ, b1 + (long)i * HID_DIM,
            mlp_g + (long)i * HID_DIM, mlp_b + (long)i * HID_DIM,
            w2pk + (long)i * WSZ, b2 + (long)i * F_DIM);
    }

    xw_kernel<<<gNC, TPB, 0, stream>>>(h, gcn_w, xw);
    deg_init_kernel<<<gN, TPB, 0, stream>>>(deg);
    deg_edge_kernel<<<gE, TPB, 0, stream>>>(dst, eattr, deg);
    dis_kernel<<<gN, TPB, 0, stream>>>(deg, dis);
    out_init_kernel<<<gNC, TPB, 0, stream>>>(xw, dis, gcn_b, out);
    gcn_edge_kernel<<<gE, TPB, 0, stream>>>(src, dst, eattr, dis, xw, out);
    lsm_kernel<<<gN, TPB, 0, stream>>>(out);
}
